// SimpleMeshEncoder_80247168959173
// MI455X (gfx1250) — matrix operand order repaired and checked by reading, not compiled, not run on hardware
//
#include <hip/hip_runtime.h>
#include <hip/hip_bf16.h>

typedef __attribute__((ext_vector_type(16))) _Float16 v16h;
typedef __attribute__((ext_vector_type(8)))  float    v8f;

#define NN 100000
#define EE 1600000
#define GG 32
#define SLOPE 0.2f

static __device__ __forceinline__ void atomicMaxF(float* addr, float val) {
  // sign-split trick -> hardware int atomics; m initialized to -inf
  if (val >= 0.0f) atomicMax((int*)addr, __float_as_int(val));
  else             atomicMin((unsigned int*)addr, __float_as_uint(val));
}

// ---------------- fill ----------------
__global__ void fill_f32(float* __restrict__ p, float v, int n) {
  int i = blockIdx.x * blockDim.x + threadIdx.x;
  if (i < n) p[i] = v;
}

// ---------------- one-shot: W [K,FOUT] f32 -> Wt [FOUT,K] f16 ----------------
__global__ void conv_w_t(const float* __restrict__ W, _Float16* __restrict__ Wt,
                         int K, int Fout) {
  int i = blockIdx.x * blockDim.x + threadIdx.x;
  if (i >= K * Fout) return;
  int k = i / Fout, f = i % Fout;
  Wt[f * K + k] = (_Float16)W[i];
}

// ---------------- layer-1 GEMM (K=3, scalar FMA) ----------------
__global__ void gemm_k3(const float* __restrict__ X, const float* __restrict__ W,
                        float* __restrict__ H, int n) {
  int i = blockIdx.x * blockDim.x + threadIdx.x;      // n*64 threads
  if (i >= n * 64) return;
  int r = i >> 6, f = i & 63;
  const float* xr = X + r * 3;
  H[i] = fmaf(xr[0], W[0 * 64 + f], fmaf(xr[1], W[1 * 64 + f], xr[2] * W[2 * 64 + f]));
}

// ---------------- WMMA GEMM: H[N,FOUT] = X[N,FIN] @ W[FIN,FOUT] ----------------
// Wt is W transposed + f16: Wt[col*FIN + k]. One wave per 16x16 output tile.
// Requires n % 16 == 0 (N = 100000 = 6250*16).
template<int FIN, int FOUT>
__global__ void gemm_wmma(const float* __restrict__ X, const _Float16* __restrict__ Wt,
                          float* __restrict__ H, int n) {
  const int lane  = threadIdx.x & 31;
  const int gwave = (blockIdx.x * blockDim.x + threadIdx.x) >> 5;
  const int colTiles = FOUT / 16;
  const int rowTile = gwave / colTiles;
  const int colTile = gwave % colTiles;
  if (rowTile * 16 >= n) return;
  const int r  = lane & 15;   // A row / B,D col within tile
  const int hh = lane >> 4;   // lane-half selector
  const int row = rowTile * 16 + r;
  const int col = colTile * 16 + r;
  v8f acc = {};
  #pragma unroll
  for (int kb = 0; kb < FIN; kb += 32) {
    v16h a, b;
    // A 16x32 f16 layout (ISA 7.12.2): lane holds row `r`,
    // K in {hh*8+0..7} u {16+hh*8+0..7} -> two contiguous 8-float chunks
    const float* xr = X + row * FIN + kb;
    #pragma unroll
    for (int j = 0; j < 8; ++j) {
      int k0 = ((j >> 2) << 4) + (hh << 3) + ((j & 3) << 1);
      a[2 * j]     = (_Float16)xr[k0];
      a[2 * j + 1] = (_Float16)xr[k0 + 1];
    }
    // B 32x16 f16 layout: lane holds col, K in {hh*8+0..7} u {16+hh*8+0..7} (same as A)
    const _Float16* wc = Wt + col * FIN + kb + (hh << 3);
    #pragma unroll
    for (int j = 0; j < 16; ++j) b[j] = wc[((j >> 3) << 4) + (j & 7)];
    acc = __builtin_amdgcn_wmma_f32_16x16x32_f16(false, a, false, b, (short)0, acc,
                                                 false, false);
  }
  // D 16x16 f32: VGPR v -> row hh*8+v, col r; n%16==0 so no per-row guard
  float* hd = H + (rowTile * 16 + (hh << 3)) * FOUT + colTile * 16 + r;
  #pragma unroll
  for (int v = 0; v < 8; ++v) hd[v * FOUT] = acc[v];
}

// ---------------- per-node attention dots: as = h.a_src, ad = h.a_dst ----------------
template<int F>
__global__ void node_dots(const float* __restrict__ H, const float* __restrict__ a_src,
                          const float* __restrict__ a_dst, float* __restrict__ as_,
                          float* __restrict__ ad_, int n) {
  int wave = (blockIdx.x * blockDim.x + threadIdx.x) >> 5;
  int lane = threadIdx.x & 31;
  if (wave >= n) return;
  float s = 0.f, d = 0.f;
  #pragma unroll
  for (int f = lane; f < F; f += 32) {
    float h = H[wave * F + f];
    s = fmaf(h, a_src[f], s);
    d = fmaf(h, a_dst[f], d);
  }
  #pragma unroll
  for (int off = 16; off > 0; off >>= 1) {
    s += __shfl_down(s, off, 32);
    d += __shfl_down(d, off, 32);
  }
  if (lane == 0) { as_[wave] = s; ad_[wave] = d; }
}

// ---------------- edge pass 1: segment max via float atomic-max ----------------
__global__ void edge_max(const int* __restrict__ src, const int* __restrict__ dst,
                         const float* __restrict__ as_, const float* __restrict__ ad_,
                         float* __restrict__ m, int ne, int n) {
  int tot = ne + n;  // virtual self-loops appended
  int i = blockIdx.x * blockDim.x + threadIdx.x;
  if (i >= tot) return;
  int s = (i < ne) ? src[i] : (i - ne);
  int d = (i < ne) ? dst[i] : (i - ne);
  float t = as_[s] + ad_[d];
  t = (t > 0.f) ? t : SLOPE * t;
  atomicMaxF(&m[d], t);
}

// ---------------- edge pass 2 (fused): denom += w; O[dst] += w*h[src] ----------------
template<int F>
__global__ void edge_agg(const int* __restrict__ src, const int* __restrict__ dst,
                         const float* __restrict__ as_, const float* __restrict__ ad_,
                         const float* __restrict__ m, const float* __restrict__ H,
                         float* __restrict__ denom, float* __restrict__ O, int ne, int n) {
  int tot  = ne + n;
  int lane = threadIdx.x & 31;
  int wave = (blockIdx.x * blockDim.x + threadIdx.x) >> 5;
  if (wave >= tot) return;
  int s = (wave < ne) ? src[wave] : (wave - ne);
  int d = (wave < ne) ? dst[wave] : (wave - ne);
  float t = as_[s] + ad_[d];
  t = (t > 0.f) ? t : SLOPE * t;
  float w = __expf(t - m[d]);
  if (lane == 0) atomicAdd(&denom[d], w);
  const float* hs = H + s * F;
  float* od = O + d * F;
  #pragma unroll
  for (int f = lane; f < F; f += 32) atomicAdd(&od[f], w * hs[f]);
}

// ---------------- per-node normalize + bias (+ReLU) ----------------
template<int F, bool RELU>
__global__ void node_final(float* __restrict__ O, const float* __restrict__ denom,
                           const float* __restrict__ b, int n) {
  int i = blockIdx.x * blockDim.x + threadIdx.x;
  if (i >= n * F) return;
  int r = i / F, f = i % F;
  float v = O[i] / denom[r] + b[f];
  if (RELU) v = (v > 0.f) ? v : 0.f;
  O[i] = v;
}

// ---------------- pooling ----------------
__global__ void pool_accum(const float* __restrict__ H, const int* __restrict__ batch,
                           float* __restrict__ sums, float* __restrict__ cnt, int n) {
  int wave = (blockIdx.x * blockDim.x + threadIdx.x) >> 5;
  int lane = threadIdx.x & 31;
  if (wave >= n) return;
  int g = batch[wave];
  if (lane == 0) atomicAdd(&cnt[g], 1.0f);
  #pragma unroll
  for (int f = lane; f < 128; f += 32) atomicAdd(&sums[g * 128 + f], H[wave * 128 + f]);
}

__global__ void pool_final(const float* __restrict__ sums, const float* __restrict__ cnt,
                           float* __restrict__ out) {
  int i = blockIdx.x * blockDim.x + threadIdx.x;
  if (i >= GG * 128) return;
  float c = cnt[i >> 7];
  out[i] = sums[i] / (c > 1.0f ? c : 1.0f);
}

static inline int cdiv(long long a, long long b) { return (int)((a + b - 1) / b); }

extern "C" void kernel_launch(void* const* d_in, const int* in_sizes, int n_in,
                              void* d_out, int out_size, void* d_ws, size_t ws_size,
                              hipStream_t stream) {
  const float* x     = (const float*)d_in[0];
  const int*   ei    = (const int*)  d_in[1];
  const int*   batch = (const int*)  d_in[2];
  const float* W1 = (const float*)d_in[3];
  const float* a1s = (const float*)d_in[4];
  const float* a1d = (const float*)d_in[5];
  const float* b1  = (const float*)d_in[6];
  const float* W2 = (const float*)d_in[7];
  const float* a2s = (const float*)d_in[8];
  const float* a2d = (const float*)d_in[9];
  const float* b2  = (const float*)d_in[10];
  const float* W3 = (const float*)d_in[11];
  const float* a3s = (const float*)d_in[12];
  const float* a3d = (const float*)d_in[13];
  const float* b3  = (const float*)d_in[14];

  const int* src = ei;        // edge_index[0,:]
  const int* dst = ei + EE;   // edge_index[1,:]

  float* ws   = (float*)d_ws;
  float* bufA = ws;                       // [N,128] : GEMM output h
  float* bufB = ws + (size_t)NN * 128;    // [N,128] : aggregation out / activation
  float* as_  = ws + (size_t)NN * 256;
  float* ad_  = as_ + NN;
  float* m_   = ad_ + NN;
  float* den_ = m_  + NN;
  float* psum = den_ + NN;                // [G,128]
  float* pcnt = psum + GG * 128;          // [G]
  _Float16* wt16 = (_Float16*)(pcnt + GG); // [<=128*64] f16 transposed weights

  const int T = 256;
  const int tot = EE + NN;  // edges + self loops

  // ---------------- Layer 1: 3 -> 64 ----------------
  gemm_k3<<<cdiv((long long)NN * 64, T), T, 0, stream>>>(x, W1, bufA, NN);
  node_dots<64><<<cdiv((long long)NN * 32, T), T, 0, stream>>>(bufA, a1s, a1d, as_, ad_, NN);
  fill_f32<<<cdiv(NN, T), T, 0, stream>>>(m_, -INFINITY, NN);
  fill_f32<<<cdiv(NN, T), T, 0, stream>>>(den_, 0.f, NN);
  fill_f32<<<cdiv((long long)NN * 64, T), T, 0, stream>>>(bufB, 0.f, NN * 64);
  edge_max<<<cdiv(tot, T), T, 0, stream>>>(src, dst, as_, ad_, m_, EE, NN);
  edge_agg<64><<<cdiv((long long)tot * 32, T), T, 0, stream>>>(src, dst, as_, ad_, m_,
                                                               bufA, den_, bufB, EE, NN);
  node_final<64, true><<<cdiv((long long)NN * 64, T), T, 0, stream>>>(bufB, den_, b1, NN);

  // ---------------- Layer 2: 64 -> 64 (WMMA) ----------------
  conv_w_t<<<cdiv(64 * 64, T), T, 0, stream>>>(W2, wt16, 64, 64);
  gemm_wmma<64, 64><<<cdiv((long long)(NN / 16) * 4 * 32, T), T, 0, stream>>>(bufB, wt16, bufA, NN);
  node_dots<64><<<cdiv((long long)NN * 32, T), T, 0, stream>>>(bufA, a2s, a2d, as_, ad_, NN);
  fill_f32<<<cdiv(NN, T), T, 0, stream>>>(m_, -INFINITY, NN);
  fill_f32<<<cdiv(NN, T), T, 0, stream>>>(den_, 0.f, NN);
  fill_f32<<<cdiv((long long)NN * 64, T), T, 0, stream>>>(bufB, 0.f, NN * 64);
  edge_max<<<cdiv(tot, T), T, 0, stream>>>(src, dst, as_, ad_, m_, EE, NN);
  edge_agg<64><<<cdiv((long long)tot * 32, T), T, 0, stream>>>(src, dst, as_, ad_, m_,
                                                               bufA, den_, bufB, EE, NN);
  node_final<64, true><<<cdiv((long long)NN * 64, T), T, 0, stream>>>(bufB, den_, b2, NN);

  // ---------------- Layer 3: 64 -> 128 (WMMA), no ReLU ----------------
  conv_w_t<<<cdiv(64 * 128, T), T, 0, stream>>>(W3, wt16, 64, 128);
  gemm_wmma<64, 128><<<cdiv((long long)(NN / 16) * 8 * 32, T), T, 0, stream>>>(bufB, wt16, bufA, NN);
  node_dots<128><<<cdiv((long long)NN * 32, T), T, 0, stream>>>(bufA, a3s, a3d, as_, ad_, NN);
  fill_f32<<<cdiv(NN, T), T, 0, stream>>>(m_, -INFINITY, NN);
  fill_f32<<<cdiv(NN, T), T, 0, stream>>>(den_, 0.f, NN);
  fill_f32<<<cdiv((long long)NN * 128, T), T, 0, stream>>>(bufB, 0.f, NN * 128);
  edge_max<<<cdiv(tot, T), T, 0, stream>>>(src, dst, as_, ad_, m_, EE, NN);
  edge_agg<128><<<cdiv((long long)tot * 32, T), T, 0, stream>>>(src, dst, as_, ad_, m_,
                                                                bufA, den_, bufB, EE, NN);
  node_final<128, false><<<cdiv((long long)NN * 128, T), T, 0, stream>>>(bufB, den_, b3, NN);

  // ---------------- global mean pool ----------------
  fill_f32<<<cdiv(GG * 128 + GG, T), T, 0, stream>>>(psum, 0.f, GG * 128 + GG);
  pool_accum<<<cdiv((long long)NN * 32, T), T, 0, stream>>>(bufB, batch, psum, pcnt, NN);
  pool_final<<<cdiv(GG * 128, T), T, 0, stream>>>(psum, pcnt, (float*)d_out);
}
